// PatentCitationMoEModule_4183298146730
// MI455X (gfx1250) — matrix operand order repaired and checked by reading, not compiled, not run on hardware
//
#include <hip/hip_runtime.h>
#include <hip/hip_bf16.h>

// ---------------------------------------------------------------------------
// Types for CDNA5 WMMA (gfx1250, wave32)
// ---------------------------------------------------------------------------
typedef __attribute__((ext_vector_type(16))) __bf16 v16bf;
typedef __attribute__((ext_vector_type(8)))  float  v8f;
typedef int v4i_async __attribute__((vector_size(16)));   // b128 payload type

#define B_TOK   2048
#define L_SEQ   64
#define D_MOD   256
#define F_BIB   512
#define H1_DIM  1024
#define H2_DIM  1024
#define O_DIM   512
#define N_EXP   8
#define N_CLS   10

__device__ __forceinline__ unsigned short bf_bits(float f) {
    unsigned u = __builtin_bit_cast(unsigned, f);
    unsigned r = u + 0x7FFFu + ((u >> 16) & 1u);   // round-to-nearest-even
    return (unsigned short)(r >> 16);
}
__device__ __forceinline__ __bf16 us2bf(unsigned short s) {
    return __builtin_bit_cast(__bf16, s);
}

// ---------------------------------------------------------------------------
// gfx1250 async global->LDS copy (16B per lane) + ASYNCcnt waits.
// Builtin signature (from toolchain diagnostic):
//   (v4i AS1* src, v4i AS3* dst, imm offset, imm cpol)
// Fallback inline CDNA5 asm:
//   global_load_async_to_lds_b128 vdst(lds byte offset), v[va:va+1], off
// ---------------------------------------------------------------------------
#if defined(__has_builtin)
#  if __has_builtin(__builtin_amdgcn_global_load_async_to_lds_b128)
#    define HAVE_ASYNC_COPY_BUILTIN 1
#  endif
#  if __has_builtin(__builtin_amdgcn_s_wait_asynccnt)
#    define HAVE_ASYNC_WAIT_BUILTIN 1
#  endif
#endif

__device__ __forceinline__ unsigned lds_byte_off(const void* p) {
    return (unsigned)(unsigned long long)
        (__attribute__((address_space(3))) const void*)p;
}

__device__ __forceinline__ void async_copy16(const void* g, void* l) {
#if defined(HAVE_ASYNC_COPY_BUILTIN)
    __builtin_amdgcn_global_load_async_to_lds_b128(
        (__attribute__((address_space(1))) v4i_async*)g,
        (__attribute__((address_space(3))) v4i_async*)l, 0, 0);
#else
    unsigned long long ga = (unsigned long long)g;
    unsigned lo = lds_byte_off(l);
    asm volatile("global_load_async_to_lds_b128 %0, %1, off"
                 :: "v"(lo), "v"(ga) : "memory");
#endif
}

__device__ __forceinline__ void wait_async3() {
#if defined(HAVE_ASYNC_WAIT_BUILTIN)
    __builtin_amdgcn_s_wait_asynccnt(3);
#else
    asm volatile("s_wait_asynccnt 0x3" ::: "memory");
#endif
}
__device__ __forceinline__ void wait_async0() {
#if defined(HAVE_ASYNC_WAIT_BUILTIN)
    __builtin_amdgcn_s_wait_asynccnt(0);
#else
    asm volatile("s_wait_asynccnt 0x0" ::: "memory");
#endif
}

// ---------------------------------------------------------------------------
// K0: q = gc@Wq + bq ; qk = Wk@q ; qb = q.bk    (scores_{bl} = x_{bl}.qk + qb)
// ---------------------------------------------------------------------------
__global__ __launch_bounds__(256) void prep_query(
    const float* __restrict__ Wq, const float* __restrict__ bq,
    const float* __restrict__ Wk, const float* __restrict__ bk,
    const float* __restrict__ gc, float* __restrict__ qkv /* [257] */) {
    __shared__ float q[D_MOD];
    __shared__ float red[D_MOD];
    int tid = threadIdx.x;
    float acc = bq[tid];
    for (int i = 0; i < D_MOD; ++i) acc += gc[i] * Wq[i * D_MOD + tid];
    q[tid] = acc;
    __syncthreads();
    float a2 = 0.f;
    for (int d = 0; d < D_MOD; ++d) a2 += Wk[tid * D_MOD + d] * q[d];
    qkv[tid] = a2;
    red[tid] = q[tid] * bk[tid];
    __syncthreads();
    for (int s = 128; s > 0; s >>= 1) {
        if (tid < s) red[tid] += red[tid + s];
        __syncthreads();
    }
    if (tid == 0) qkv[D_MOD] = red[0];
}

// ---------------------------------------------------------------------------
// K1: per batch row: gather x = ipc_emb[ii]+role_emb[ri], masked softmax of
//     x.qk + qb over L, then xp[b] = sum_l attn_l * x_l
// ---------------------------------------------------------------------------
__global__ __launch_bounds__(256) void attn_pool(
    const int* __restrict__ ipc_idx, const int* __restrict__ role_idx,
    const int* __restrict__ mask,
    const float* __restrict__ ipc_emb, const float* __restrict__ role_emb,
    const float* __restrict__ qkv, float* __restrict__ xp) {
    __shared__ int   ii[L_SEQ], ri[L_SEQ], mk[L_SEQ];
    __shared__ float sqk[D_MOD];
    __shared__ float part[256];
    __shared__ float sc[L_SEQ];
    __shared__ float at[L_SEQ];
    int b = blockIdx.x, tid = threadIdx.x;
    sqk[tid] = qkv[tid];
    if (tid < L_SEQ) {
        ii[tid] = ipc_idx[b * L_SEQ + tid];
        ri[tid] = role_idx[b * L_SEQ + tid];
        mk[tid] = mask[b * L_SEQ + tid];
    }
    __syncthreads();
    int t = tid >> 2, q0 = (tid & 3) * 64;
    const float* ip = ipc_emb + (size_t)ii[t] * D_MOD + q0;
    const float* rp = role_emb + (size_t)ri[t] * D_MOD + q0;
    float p = 0.f;
    for (int d = 0; d < 64; ++d) p += (ip[d] + rp[d]) * sqk[q0 + d];
    part[tid] = p;
    __syncthreads();
    if (tid < L_SEQ) {
        float s = part[tid * 4] + part[tid * 4 + 1] + part[tid * 4 + 2] +
                  part[tid * 4 + 3] + qkv[D_MOD];
        sc[tid] = (mk[tid] == 0) ? -1e9f : s;
    }
    __syncthreads();
    float mx = -1e30f;
    for (int j = 0; j < L_SEQ; ++j) mx = fmaxf(mx, sc[j]);
    if (tid < L_SEQ) at[tid] = __expf(sc[tid] - mx);
    __syncthreads();
    float sum = 0.f;
    for (int j = 0; j < L_SEQ; ++j) sum += at[j];
    float inv = 1.f / sum;
    float acc = 0.f;
    for (int j = 0; j < L_SEQ; ++j) {
        float w = at[j] * inv;
        acc += w * (ipc_emb[(size_t)ii[j] * D_MOD + tid] +
                    role_emb[(size_t)ri[j] * D_MOD + tid]);
    }
    xp[(size_t)b * D_MOD + tid] = acc;
}

// ---------------------------------------------------------------------------
// K2: pooled = xp@Wv + bv ; LayerNorm ; affinity = rin@gate_W + gate_b ;
//     top-2 on (affinity + expert_biases), gates softmaxed on unbiased affinity
// ---------------------------------------------------------------------------
__global__ __launch_bounds__(256) void route(
    const float* __restrict__ xp, const float* __restrict__ Wv,
    const float* __restrict__ bv, const float* __restrict__ ln_g,
    const float* __restrict__ ln_b, const float* __restrict__ gate_W,
    const float* __restrict__ gate_b, const float* __restrict__ ebias,
    float* __restrict__ gw_out, int* __restrict__ sel_out,
    int* __restrict__ idx_out) {
    __shared__ float sx[D_MOD], pl[D_MOD], rn[D_MOD], aff[N_EXP], selv[N_EXP];
    int b = blockIdx.x, tid = threadIdx.x;
    sx[tid] = xp[(size_t)b * D_MOD + tid];
    __syncthreads();
    float acc = bv[tid];
    for (int i = 0; i < D_MOD; ++i) acc += sx[i] * Wv[i * D_MOD + tid];
    pl[tid] = acc;
    __syncthreads();
    float mu = 0.f;
    for (int i = 0; i < D_MOD; ++i) mu += pl[i];
    mu *= (1.f / D_MOD);
    float var = 0.f;
    for (int i = 0; i < D_MOD; ++i) { float d = pl[i] - mu; var += d * d; }
    var *= (1.f / D_MOD);
    float rstd = rsqrtf(var + 1e-5f);
    rn[tid] = (pl[tid] - mu) * rstd * ln_g[tid] + ln_b[tid];
    __syncthreads();
    if (tid < N_EXP) {
        float a = gate_b[tid];
        for (int d = 0; d < D_MOD; ++d) a += rn[d] * gate_W[d * N_EXP + tid];
        aff[tid]  = a;
        selv[tid] = a + ebias[tid];
    }
    __syncthreads();
    if (tid == 0) {
        int i1 = 0; float v1 = selv[0];
        for (int e = 1; e < N_EXP; ++e) if (selv[e] > v1) { v1 = selv[e]; i1 = e; }
        int i2 = -1; float v2 = -1e30f;
        for (int e = 0; e < N_EXP; ++e)
            if (e != i1 && selv[e] > v2) { v2 = selv[e]; i2 = e; }
        float a1 = aff[i1], a2 = aff[i2];
        float m = fmaxf(a1, a2);
        float e1 = __expf(a1 - m), e2 = __expf(a2 - m);
        float inv = 1.f / (e1 + e2);
        gw_out[b * 2] = e1 * inv;  gw_out[b * 2 + 1] = e2 * inv;
        sel_out[b * 2] = i1;       sel_out[b * 2 + 1] = i2;
        idx_out[b * 2] = i1;       idx_out[b * 2 + 1] = i2;
    }
}

// ---------------------------------------------------------------------------
// Weight / activation preparation: fp32 -> bf16 (optionally transposed [K,N]
// -> [N,K] so GEMM B-tiles are contiguous rows for async B128 copies)
// ---------------------------------------------------------------------------
__global__ __launch_bounds__(256) void cast_bf16(
    const float* __restrict__ src, unsigned short* __restrict__ dst, int n) {
    int i = blockIdx.x * 256 + threadIdx.x;
    if (i < n) dst[i] = bf_bits(src[i]);
}
__global__ __launch_bounds__(256) void cast_transpose_bf16(
    const float* __restrict__ src, unsigned short* __restrict__ dst,
    int K, int N) {
    int i = blockIdx.x * 256 + threadIdx.x;
    if (i < K * N) {
        int k = i / N, n = i % N;
        dst[(size_t)n * K + k] = bf_bits(src[i]);   // coalesced read
    }
}

// ---------------------------------------------------------------------------
// WMMA bf16 GEMM: C[M,N] = act(A[M,K] @ Bt[N,K]^T + bias)
//   block 128x64, BK=32; 8 wave32s, each wave a 32x32 tile (2x2 WMMA frags).
//   Tiles staged with async global->LDS B128 copies, double-buffered on
//   ASYNCcnt (3 async instructions per wave per tile -> s_wait_asynccnt 3).
// ---------------------------------------------------------------------------
#define BM 128
#define BN 64
#define BK 32
#define LDT (BK + 8)   // row stride 80B: 16B-aligned + conflict-free frag reads

template <bool RELU, bool OUT_BF16>
__global__ __launch_bounds__(256) void gemm_bf16_wmma(
    const unsigned short* __restrict__ A,   // [M,K] bf16
    const unsigned short* __restrict__ Bt,  // [N,K] bf16 (pre-transposed)
    const float* __restrict__ bias, void* __restrict__ Cptr,
    int M, int N, int K) {
    __shared__ __align__(16) unsigned short As[2][BM * LDT];
    __shared__ __align__(16) unsigned short Bs[2][BN * LDT];

    const int tid  = threadIdx.x;
    const int lane = tid & 31;
    const int wave = tid >> 5;
    const int wm   = wave & 3;     // 4 waves along M
    const int wn   = wave >> 2;    // 2 waves along N
    const int bm0  = blockIdx.y * BM;
    const int bn0  = blockIdx.x * BN;

    v8f acc[2][2];
    for (int i = 0; i < 2; ++i)
        for (int j = 0; j < 2; ++j)
            for (int q = 0; q < 8; ++q) acc[i][j][q] = 0.f;

    const int NK = K / BK;

    // ---- async stage of one K-tile into buffer `buf` (3 asyncs per wave) ----
    auto stage = [&](int kt, int buf) {
        const int k0 = kt * BK;
        // A tile: 128x32 bf16 = 512 x 16B chunks, 2 per thread
        for (int c = tid; c < (BM * BK) / 8; c += 256) {
            int r = c >> 2, cc = c & 3;           // 4 chunks per row
            const void* g = (const char*)A +
                (((size_t)(bm0 + r) * K + k0) << 1) + cc * 16;
            void* l = (char*)&As[buf][r * LDT] + cc * 16;
            async_copy16(g, l);
        }
        // B tile: 64x32 bf16 = 256 x 16B chunks, 1 per thread
        {
            int n = tid >> 2, cc = tid & 3;
            const void* g = (const char*)Bt +
                (((size_t)(bn0 + n) * K + k0) << 1) + cc * 16;
            void* l = (char*)&Bs[buf][n * LDT] + cc * 16;
            async_copy16(g, l);
        }
    };

    stage(0, 0);
    for (int kt = 0; kt < NK; ++kt) {
        const int cur = kt & 1;
        if (kt + 1 < NK) {
            stage(kt + 1, cur ^ 1);
            wait_async3();            // drain the 3 copies of the current tile
        } else {
            wait_async0();
        }
        __syncthreads();

        const unsigned short* Asc = As[cur];
        const unsigned short* Bsc = Bs[cur];

        // A fragments: 16-bit A layout (lane>=16 -> K+8; vgpr j>=4 -> K+16)
        v16bf afr[2], bfr[2];
        const int khalfA = (lane >> 4) ? 8 : 0;
        for (int ti = 0; ti < 2; ++ti) {
            int m = wm * 32 + ti * 16 + (lane & 15);
            for (int j = 0; j < 8; ++j) {
                int k = ((j & 3) << 1) + ((j >> 2) << 4) + khalfA;
                afr[ti][2 * j]     = us2bf(Asc[m * LDT + k]);
                afr[ti][2 * j + 1] = us2bf(Asc[m * LDT + k + 1]);
            }
        }
        // B fragments: same 16-bit layout as A (lane>=16 -> K+8; vgpr j>=4 -> K+16)
        const int khalfB = (lane >> 4) ? 8 : 0;
        for (int tj = 0; tj < 2; ++tj) {
            int n = wn * 32 + tj * 16 + (lane & 15);
            for (int j = 0; j < 8; ++j) {
                int k = ((j & 3) << 1) + ((j >> 2) << 4) + khalfB;
                bfr[tj][2 * j]     = us2bf(Bsc[n * LDT + k]);
                bfr[tj][2 * j + 1] = us2bf(Bsc[n * LDT + k + 1]);
            }
        }
        for (int ti = 0; ti < 2; ++ti)
            for (int tj = 0; tj < 2; ++tj)
                acc[ti][tj] = __builtin_amdgcn_wmma_f32_16x16x32_bf16(
                    false, afr[ti], false, bfr[tj], (short)0, acc[ti][tj],
                    false, false);
        __syncthreads();
    }

    // epilogue: bias (+ReLU), write bf16 or fp32
    for (int ti = 0; ti < 2; ++ti) {
        int row_base = bm0 + wm * 32 + ti * 16 + ((lane >> 4) ? 8 : 0);
        for (int tj = 0; tj < 2; ++tj) {
            int ncol = bn0 + wn * 32 + tj * 16 + (lane & 15);
            float bvv = bias[ncol];
            for (int j = 0; j < 8; ++j) {
                int row = row_base + j;
                float v = acc[ti][tj][j] + bvv;
                if (RELU) v = fmaxf(v, 0.f);
                if (OUT_BF16)
                    ((unsigned short*)Cptr)[(size_t)row * N + ncol] = bf_bits(v);
                else
                    ((float*)Cptr)[(size_t)row * N + ncol] = v;
            }
        }
    }
}

// ---------------------------------------------------------------------------
// K-final: moe_out = g1*EO[i1,b] + g2*EO[i2,b] ; logits = moe_out@head_W + b
// ---------------------------------------------------------------------------
__global__ __launch_bounds__(256) void combine_head(
    const float* __restrict__ EO, const float* __restrict__ gw,
    const int* __restrict__ sel, const float* __restrict__ head_W,
    const float* __restrict__ head_b, float* __restrict__ logits) {
    __shared__ float mo[O_DIM];
    int b = blockIdx.x, tid = threadIdx.x;
    int i1 = sel[b * 2], i2 = sel[b * 2 + 1];
    float g1 = gw[b * 2], g2 = gw[b * 2 + 1];
    const float* e1 = EO + ((size_t)i1 * B_TOK + b) * O_DIM;
    const float* e2 = EO + ((size_t)i2 * B_TOK + b) * O_DIM;
    mo[tid]       = g1 * e1[tid]       + g2 * e2[tid];
    mo[tid + 256] = g1 * e1[tid + 256] + g2 * e2[tid + 256];
    __syncthreads();
    if (tid < N_CLS) {
        float a = head_b[tid];
        for (int o = 0; o < O_DIM; ++o) a += mo[o] * head_W[o * N_CLS + tid];
        logits[b * N_CLS + tid] = a;
    }
}

// ---------------------------------------------------------------------------
// launch
// ---------------------------------------------------------------------------
static inline size_t align_up(size_t x, size_t a) { return (x + a - 1) & ~(a - 1); }

extern "C" void kernel_launch(void* const* d_in, const int* in_sizes, int n_in,
                              void* d_out, int out_size, void* d_ws, size_t ws_size,
                              hipStream_t stream) {
    const int*   ipc_idx  = (const int*)d_in[0];
    const int*   role_idx = (const int*)d_in[1];
    const float* bib      = (const float*)d_in[2];
    const int*   pmask    = (const int*)d_in[3];
    // d_in[4] = top_k (== 2, hardcoded)
    const float* ipc_emb  = (const float*)d_in[5];
    const float* role_emb = (const float*)d_in[6];
    const float* Wq = (const float*)d_in[7];  const float* bq = (const float*)d_in[8];
    const float* Wk = (const float*)d_in[9];  const float* bk = (const float*)d_in[10];
    const float* Wv = (const float*)d_in[11]; const float* bv = (const float*)d_in[12];
    const float* gc = (const float*)d_in[13];
    const float* ln_g = (const float*)d_in[14]; const float* ln_b = (const float*)d_in[15];
    const float* gate_W = (const float*)d_in[16]; const float* gate_b = (const float*)d_in[17];
    const float* ebias = (const float*)d_in[18];
    const float* W1 = (const float*)d_in[19]; const float* b1 = (const float*)d_in[20];
    const float* W2 = (const float*)d_in[21]; const float* b2 = (const float*)d_in[22];
    const float* W3 = (const float*)d_in[23]; const float* b3 = (const float*)d_in[24];
    const float* head_W = (const float*)d_in[25]; const float* head_b = (const float*)d_in[26];

    // workspace carve-up (~50 MB)
    char* w = (char*)d_ws;
    float* qkv = (float*)w;                  w += align_up(257 * sizeof(float), 256);
    float* xp  = (float*)w;                  w += align_up((size_t)B_TOK * D_MOD * 4, 256);
    float* gw  = (float*)w;                  w += align_up((size_t)B_TOK * 2 * 4, 256);
    int*   sel = (int*)w;                    w += align_up((size_t)B_TOK * 2 * 4, 256);
    unsigned short* bibB = (unsigned short*)w; w += align_up((size_t)B_TOK * F_BIB * 2, 256);
    unsigned short* W1t = (unsigned short*)w;  w += align_up((size_t)H1_DIM * F_BIB * 2, 256);
    unsigned short* W2t = (unsigned short*)w;  w += align_up((size_t)H2_DIM * H1_DIM * 2, 256);
    unsigned short* W3t = (unsigned short*)w;  w += align_up((size_t)O_DIM * H2_DIM * 2, 256);
    unsigned short* H1 = (unsigned short*)w;   w += align_up((size_t)B_TOK * H1_DIM * 2, 256);
    unsigned short* H2 = (unsigned short*)w;   w += align_up((size_t)B_TOK * H2_DIM * 2, 256);
    float* EO = (float*)w;                   w += align_up((size_t)N_EXP * B_TOK * O_DIM * 4, 256);

    float* logits  = (float*)d_out;
    int*   idx_out = (int*)((float*)d_out + (size_t)B_TOK * N_CLS);

    prep_query<<<1, 256, 0, stream>>>(Wq, bq, Wk, bk, gc, qkv);
    attn_pool<<<B_TOK, 256, 0, stream>>>(ipc_idx, role_idx, pmask,
                                         ipc_emb, role_emb, qkv, xp);
    route<<<B_TOK, 256, 0, stream>>>(xp, Wv, bv, ln_g, ln_b, gate_W, gate_b,
                                     ebias, gw, sel, idx_out);

    cast_bf16<<<(B_TOK * F_BIB + 255) / 256, 256, 0, stream>>>(
        bib, bibB, B_TOK * F_BIB);

    for (int e = 0; e < N_EXP; ++e) {
        cast_transpose_bf16<<<(F_BIB * H1_DIM + 255) / 256, 256, 0, stream>>>(
            W1 + (size_t)e * F_BIB * H1_DIM, W1t, F_BIB, H1_DIM);
        gemm_bf16_wmma<true, true>
            <<<dim3(H1_DIM / BN, B_TOK / BM), 256, 0, stream>>>(
                bibB, W1t, b1 + (size_t)e * H1_DIM, H1, B_TOK, H1_DIM, F_BIB);

        cast_transpose_bf16<<<(H1_DIM * H2_DIM + 255) / 256, 256, 0, stream>>>(
            W2 + (size_t)e * H1_DIM * H2_DIM, W2t, H1_DIM, H2_DIM);
        gemm_bf16_wmma<true, true>
            <<<dim3(H2_DIM / BN, B_TOK / BM), 256, 0, stream>>>(
                H1, W2t, b2 + (size_t)e * H2_DIM, H2, B_TOK, H2_DIM, H1_DIM);

        cast_transpose_bf16<<<(H2_DIM * O_DIM + 255) / 256, 256, 0, stream>>>(
            W3 + (size_t)e * H2_DIM * O_DIM, W3t, H2_DIM, O_DIM);
        gemm_bf16_wmma<false, false>
            <<<dim3(O_DIM / BN, B_TOK / BM), 256, 0, stream>>>(
                H2, W3t, b3 + (size_t)e * O_DIM,
                EO + (size_t)e * B_TOK * O_DIM, B_TOK, O_DIM, H2_DIM);
    }

    combine_head<<<B_TOK, 256, 0, stream>>>(EO, gw, sel, head_W, head_b, logits);
}
